// TransformerBlock_28930899705911
// MI455X (gfx1250) — matrix operand order repaired and checked by reading, not compiled, not run on hardware
//
#include <hip/hip_runtime.h>
#include <cstdint>
#include <cstddef>

#define DEV __device__ __forceinline__

constexpr int kB = 2, kL = 2048, kD = 1024, kH = 16, kHD = 64, kFF = 4096;
constexpr float kEps = 1e-5f;

typedef __bf16 bf16;
typedef __attribute__((ext_vector_type(16))) __bf16 bf16x16;
typedef __attribute__((ext_vector_type(8)))  __bf16 bf16x8;
typedef __attribute__((ext_vector_type(8)))  float  f32x8;

// Build a 16-element bf16 A/B fragment from two 16-byte LDS/global chunks.
DEV bf16x16 mk16(const bf16* lo, const bf16* hi) {
  bf16x8 a = *(const bf16x8*)lo;
  bf16x8 b = *(const bf16x8*)hi;
  bf16x16 r;
#pragma unroll
  for (int i = 0; i < 8; ++i) { r[i] = a[i]; r[i + 8] = b[i]; }
  return r;
}

DEV f32x8 wmma_bf16(bf16x16 a, bf16x16 b, f32x8 c) {
  // D = A(16x32) * B(32x16) + C, f32 accumulate
  return __builtin_amdgcn_wmma_f32_16x16x32_bf16(
      false, a, false, b, (short)0, c, false, false);
}

// Async copy of one 16-byte chunk global -> LDS (per-lane), ASYNCcnt-tracked.
DEV void async_b128(uint32_t lds_off, const void* gptr) {
  asm volatile("global_load_async_to_lds_b128 %0, %1, off"
               :: "v"(lds_off), "v"((uint64_t)(uintptr_t)gptr) : "memory");
}
// Async loads retire in order: waiting <=8 releases the older tile's 8 chunks
// while the newer tile's 8 are still in flight (double-buffer pipeline).
DEV void wait_async8() { asm volatile("s_wait_asynccnt 0x8" ::: "memory"); }
DEV void wait_async0() { asm volatile("s_wait_asynccnt 0x0" ::: "memory"); }

// ---------------------------------------------------------------------------
// LayerNorm: one 256-thread block per row of D=1024, bf16 output
// ---------------------------------------------------------------------------
__global__ __launch_bounds__(256) void layernorm_kernel(
    const float* __restrict__ x, const float* __restrict__ g,
    const float* __restrict__ b, bf16* __restrict__ out) {
  __shared__ float s1[256], s2[256];
  const int tid = threadIdx.x;
  const float* row = x + (size_t)blockIdx.x * kD;
  float a = 0.f, q = 0.f;
  for (int i = tid; i < kD; i += 256) { float v = row[i]; a += v; q += v * v; }
  s1[tid] = a; s2[tid] = q;
  __syncthreads();
  for (int off = 128; off > 0; off >>= 1) {
    if (tid < off) { s1[tid] += s1[tid + off]; s2[tid] += s2[tid + off]; }
    __syncthreads();
  }
  const float mean = s1[0] * (1.f / kD);
  const float var = s2[0] * (1.f / kD) - mean * mean;
  const float rs = rsqrtf(var + kEps);
  bf16* orow = out + (size_t)blockIdx.x * kD;
  for (int i = tid; i < kD; i += 256)
    orow[i] = (bf16)((row[i] - mean) * rs * g[i] + b[i]);
}

// ---------------------------------------------------------------------------
// Weight transpose+convert: W[K,N] f32 -> Wt[N,K] bf16 (32x32 LDS tiles).
// ---------------------------------------------------------------------------
__global__ __launch_bounds__(256) void transpose_w_kernel(
    const float* __restrict__ W, bf16* __restrict__ Wt, int K, int N) {
  __shared__ bf16 t[32][33];
  const int k0 = blockIdx.x * 32, n0 = blockIdx.y * 32;
  const int tx = threadIdx.x & 31, ty = threadIdx.x >> 5;  // ty in 0..7
#pragma unroll
  for (int p = 0; p < 4; ++p) {
    int k = ty + p * 8;
    t[tx][k] = (bf16)W[(size_t)(k0 + k) * N + n0 + tx];
  }
  __syncthreads();
#pragma unroll
  for (int p = 0; p < 4; ++p) {
    int n = ty + p * 8;
    Wt[(size_t)(n0 + n) * K + k0 + tx] = t[n][tx];
  }
}

// ---------------------------------------------------------------------------
// WMMA GEMM: C[M,N] = A[M,K](bf16) * Wt[N,K](bf16)^T + bias (+epilogue)
// Block tile 128x128, BK=64. 8 waves in 4x2; each wave owns 32x64 =
// 2x4 WMMA tiles (16 v_wmma per k-step). Double-buffered LDS staging with
// GLOBAL_LOAD_ASYNC_TO_LDS_B128: tile t+1 streams in (ASYNCcnt) while the
// WMMA pipe consumes tile t.
// ---------------------------------------------------------------------------
enum { MODE_BF16 = 0, MODE_GELU = 1, MODE_RESID = 2 };

template <int MODE>
__global__ __launch_bounds__(256) void gemm_wmma(
    const bf16* __restrict__ A, const bf16* __restrict__ Wt,
    const float* __restrict__ bias, const float* __restrict__ resid,
    bf16* __restrict__ Obf, float* __restrict__ Of, int M, int N, int K) {
  __shared__ __align__(16) bf16 sA[2][128][64];  // 2 x 16 KB, m-major
  __shared__ __align__(16) bf16 sB[2][128][64];  // 2 x 16 KB, n-major
  const int tid = threadIdx.x;
  const int bm = blockIdx.y * 128, bn = blockIdx.x * 128;
  const int wave = tid >> 5, lane = tid & 31;
  const int wm = (wave & 3) * 32, wn = (wave >> 2) * 64;
  const int lm = lane & 15;
  const int ka = (lane < 16) ? 0 : 8;    // A-frag K base (16-bit A layout)
  const int kb_ = (lane < 16) ? 0 : 8;   // B-frag K base (16-bit B layout)
  f32x8 acc[2][4] = {};

  const uint32_t ldsA0 = (uint32_t)(uintptr_t)(&sA[0][0][0]);
  const uint32_t ldsB0 = (uint32_t)(uintptr_t)(&sB[0][0][0]);
  constexpr uint32_t kBufBytes = 128 * 64 * 2;  // 16 KB per buffer

  // Issue the 8 async b128 chunks (4 A + 4 B) for tile t into buffer t&1.
  auto issue = [&](int t) {
    const int k0 = t * 64;
    const uint32_t bufo = (uint32_t)(t & 1) * kBufBytes;
#pragma unroll
    for (int i = 0; i < 4; ++i) {
      const int idx = tid + i * 256;                 // 0..1023
      const int r = idx >> 3, c = (idx & 7) * 8;
      const uint32_t lo = bufo + (uint32_t)(r * 128 + c * 2);
      async_b128(ldsA0 + lo, A + (size_t)(bm + r) * K + k0 + c);
      async_b128(ldsB0 + lo, Wt + (size_t)(bn + r) * K + k0 + c);
    }
  };

  const int T = K >> 6;  // number of BK=64 tiles
  issue(0);
  for (int t = 0; t < T; ++t) {
    if (t + 1 < T) {
      issue(t + 1);    // prefetch next tile into the other buffer
      wait_async8();   // tile t complete (in-order retirement), t+1 in flight
    } else {
      wait_async0();
    }
    __syncthreads();

    const bf16(*cA)[64] = sA[t & 1];
    const bf16(*cB)[64] = sB[t & 1];
#pragma unroll
    for (int ks = 0; ks < 2; ++ks) {
      bf16x16 af[2], bfr[4];
#pragma unroll
      for (int mi = 0; mi < 2; ++mi) {
        const bf16* p = &cA[wm + mi * 16 + lm][ks * 32 + ka];
        af[mi] = mk16(p, p + 16);       // K 0..7,16..23 / 8..15,24..31
      }
#pragma unroll
      for (int ni = 0; ni < 4; ++ni) {
        const bf16* p = &cB[wn + ni * 16 + lm][ks * 32 + kb_];
        bfr[ni] = mk16(p, p + 16);      // K 0..7,16..23 / 8..15,24..31
      }
#pragma unroll
      for (int mi = 0; mi < 2; ++mi)
#pragma unroll
        for (int ni = 0; ni < 4; ++ni)
          acc[mi][ni] = wmma_bf16(af[mi], bfr[ni], acc[mi][ni]);
    }
    __syncthreads();  // all waves done with buffer t&1 before tile t+2 issues
  }

  // Epilogue. C layout: lane holds column N=lm, rows r + 8*(lane>=16).
  const int half8 = (lane >> 4) * 8;
#pragma unroll
  for (int mi = 0; mi < 2; ++mi)
#pragma unroll
    for (int ni = 0; ni < 4; ++ni) {
      const int col = bn + wn + ni * 16 + lm;
      const float bv = bias[col];
#pragma unroll
      for (int r = 0; r < 8; ++r) {
        const int row = bm + wm + mi * 16 + half8 + r;
        float v = acc[mi][ni][r] + bv;
        if constexpr (MODE == MODE_GELU)
          v = 0.5f * v * (1.0f + erff(v * 0.70710678118f));
        if constexpr (MODE == MODE_RESID)
          Of[(size_t)row * N + col] = v + resid[(size_t)row * N + col];
        else
          Obf[(size_t)row * N + col] = (bf16)v;
      }
    }
}

// ---------------------------------------------------------------------------
// RoPE + attention-layout split.
// qkv: [B*L, 3*D] bf16.  Q,K out: [B,H,L,HD] bf16 (roped).
// V out transposed: [B,H,HD,L] bf16 (so PV B-fragments are contiguous).
// ---------------------------------------------------------------------------
__global__ __launch_bounds__(256) void rope_split_kernel(
    const bf16* __restrict__ qkv, bf16* __restrict__ Q, bf16* __restrict__ Kc,
    bf16* __restrict__ Vt) {
  const int idx = blockIdx.x * 256 + threadIdx.x;
  const int i = idx & 31;
  const int h = (idx >> 5) & (kH - 1);
  const int l = (idx >> 9) & (kL - 1);
  const int b = idx >> 20;
  const size_t base = ((size_t)(b * kL + l)) * (3 * kD) + h * kHD;
  const float inv = exp2f(-(float)(2 * i) * (13.287712379549449f / 64.f));
  float s, c;
  __sincosf((float)l * inv, &s, &c);
  const size_t ro = ((size_t)(b * kH + h) * kL + l) * kHD;
  {
    float t1 = (float)qkv[base + i], t2 = (float)qkv[base + 32 + i];
    Q[ro + i]      = (bf16)(t1 * c - t2 * s);
    Q[ro + 32 + i] = (bf16)(t1 * s + t2 * c);
  }
  {
    float t1 = (float)qkv[base + kD + i], t2 = (float)qkv[base + kD + 32 + i];
    Kc[ro + i]      = (bf16)(t1 * c - t2 * s);
    Kc[ro + 32 + i] = (bf16)(t1 * s + t2 * c);
  }
  {
    const size_t vo = (size_t)(b * kH + h) * kHD;
    Vt[(vo + i) * kL + l]      = qkv[base + 2 * kD + i];
    Vt[(vo + 32 + i) * kL + l] = qkv[base + 2 * kD + 32 + i];
  }
}

// ---------------------------------------------------------------------------
// Causal flash attention. Grid (L/128, H, B), 256 threads = 8 waves.
// Each wave owns 16 query rows; loops key blocks of 32 with online softmax.
// ---------------------------------------------------------------------------
__global__ __launch_bounds__(256) void attn_kernel(
    const bf16* __restrict__ Q, const bf16* __restrict__ Kc,
    const bf16* __restrict__ Vt, bf16* __restrict__ O) {
  __shared__ __align__(16) bf16 sP[8][16][32];  // per-wave P bounce buffer
  const int wave = threadIdx.x >> 5, lane = threadIdx.x & 31;
  const int b = blockIdx.z, h = blockIdx.y;
  const int q0 = blockIdx.x * 128 + wave * 16;
  const bf16* Qh = Q  + (size_t)(b * kH + h) * kL * kHD;
  const bf16* Kh = Kc + (size_t)(b * kH + h) * kL * kHD;
  const bf16* Vh = Vt + (size_t)(b * kH + h) * kHD * kL;
  const int lm = lane & 15;
  const int half = lane >> 4;
  const int ka = half ? 8 : 0;
  const int kb16 = half ? 8 : 0;

  bf16x16 qf[2];
#pragma unroll
  for (int cc = 0; cc < 2; ++cc) {
    const bf16* p = Qh + (size_t)(q0 + lm) * kHD + cc * 32 + ka;
    qf[cc] = mk16(p, p + 16);
  }
  f32x8 o[4] = {};
  float mrow[8], lrow[8];
#pragma unroll
  for (int r = 0; r < 8; ++r) { mrow[r] = -3.0e38f; lrow[r] = 0.f; }

  bf16* myP = &sP[wave][0][0];
  const int kend = q0 + 16;  // causal: keys 0..q0+15
  for (int kb = 0; kb < kend; kb += 32) {
    f32x8 s[2] = {};
#pragma unroll
    for (int t = 0; t < 2; ++t)
#pragma unroll
      for (int cc = 0; cc < 2; ++cc) {
        const bf16* p = Kh + (size_t)(kb + t * 16 + lm) * kHD + cc * 32 + kb16;
        s[t] = wmma_bf16(qf[cc], mk16(p, p + 16), s[t]);
      }
    float p0a[8], p1a[8];
#pragma unroll
    for (int r = 0; r < 8; ++r) {
      const int qrow = q0 + half * 8 + r;
      float v0 = s[0][r] * 0.125f;          // HD^-0.5
      float v1 = s[1][r] * 0.125f;
      if (kb + lm > qrow) v0 = -3.0e38f;
      if (kb + 16 + lm > qrow) v1 = -3.0e38f;
      float mx = fmaxf(v0, v1);
      mx = fmaxf(mx, __shfl_xor(mx, 1));
      mx = fmaxf(mx, __shfl_xor(mx, 2));
      mx = fmaxf(mx, __shfl_xor(mx, 4));
      mx = fmaxf(mx, __shfl_xor(mx, 8));
      const float mnew = fmaxf(mrow[r], mx);
      const float corr = __expf(mrow[r] - mnew);
      const float p0 = __expf(v0 - mnew);
      const float p1 = __expf(v1 - mnew);
      float psum = p0 + p1;
      psum += __shfl_xor(psum, 1);
      psum += __shfl_xor(psum, 2);
      psum += __shfl_xor(psum, 4);
      psum += __shfl_xor(psum, 8);
      lrow[r] = lrow[r] * corr + psum;
      mrow[r] = mnew;
#pragma unroll
      for (int cc = 0; cc < 4; ++cc) o[cc][r] *= corr;
      p0a[r] = p0; p1a[r] = p1;
    }
    // C-layout -> A-layout conversion through wave-private LDS
#pragma unroll
    for (int r = 0; r < 8; ++r) {
      myP[(half * 8 + r) * 32 + lm]      = (bf16)p0a[r];
      myP[(half * 8 + r) * 32 + 16 + lm] = (bf16)p1a[r];
    }
    asm volatile("s_wait_dscnt 0x0" ::: "memory");
    bf16x16 pf;
    {
      const bf16* p = myP + lm * 32 + ka;
      pf = mk16(p, p + 16);
    }
#pragma unroll
    for (int cc = 0; cc < 4; ++cc) {
      const bf16* p = Vh + (size_t)(cc * 16 + lm) * kL + kb + kb16;
      o[cc] = wmma_bf16(pf, mk16(p, p + 16), o[cc]);
    }
  }
#pragma unroll
  for (int cc = 0; cc < 4; ++cc)
#pragma unroll
    for (int r = 0; r < 8; ++r) {
      const int qrow = q0 + half * 8 + r;
      O[((size_t)(b * kL + qrow)) * kD + h * kHD + cc * 16 + lm] =
          (bf16)(o[cc][r] / lrow[r]);
    }
}

// ---------------------------------------------------------------------------
// Host side
// ---------------------------------------------------------------------------
extern "C" void kernel_launch(void* const* d_in, const int* in_sizes, int n_in,
                              void* d_out, int out_size, void* d_ws, size_t ws_size,
                              hipStream_t stream) {
  const float* x      = (const float*)d_in[0];
  const float* qkv_w  = (const float*)d_in[1];
  const float* qkv_b  = (const float*)d_in[2];
  const float* proj_w = (const float*)d_in[3];
  const float* proj_b = (const float*)d_in[4];
  const float* ff1_w  = (const float*)d_in[5];
  const float* ff1_b  = (const float*)d_in[6];
  const float* ff2_w  = (const float*)d_in[7];
  const float* ff2_b  = (const float*)d_in[8];
  const float* ln1_g  = (const float*)d_in[9];
  const float* ln1_b  = (const float*)d_in[10];
  const float* ln2_g  = (const float*)d_in[11];
  const float* ln2_b  = (const float*)d_in[12];

  char* ws = (char*)d_ws;
  const size_t MB = (size_t)1 << 20;
  bf16*  xn1   = (bf16*)(ws + 0 * MB);     // [M,D]    bf16  (8 MB)
  bf16*  qkv   = (bf16*)(ws + 8 * MB);     // [M,3D]   bf16  (24 MB)
  bf16*  Qb    = (bf16*)(ws + 32 * MB);    // [B,H,L,HD]     (8 MB)
  bf16*  Kb    = (bf16*)(ws + 40 * MB);    // [B,H,L,HD]     (8 MB)
  bf16*  Vt    = (bf16*)(ws + 48 * MB);    // [B,H,HD,L]     (8 MB)
  bf16*  attn  = (bf16*)(ws + 56 * MB);    // [M,D]    bf16  (8 MB)
  float* x2    = (float*)(ws + 64 * MB);   // [M,D]    f32   (16 MB)
  bf16*  xn2   = (bf16*)(ws + 80 * MB);    // [M,D]    bf16  (8 MB)
  bf16*  hact  = (bf16*)(ws + 88 * MB);    // [M,FF]   bf16  (32 MB)
  bf16*  qkvWt = (bf16*)(ws + 120 * MB);   // [3D,D]   bf16  (6 MB)
  bf16*  prjWt = (bf16*)(ws + 126 * MB);   // [D,D]    bf16  (2 MB)
  bf16*  ff1Wt = (bf16*)(ws + 128 * MB);   // [FF,D]   bf16  (8 MB)
  bf16*  ff2Wt = (bf16*)(ws + 136 * MB);   // [D,FF]   bf16  (8 MB)
  float* out   = (float*)d_out;

  const int M = kB * kL;  // 4096

  // 0. One-time (per launch) weight transpose+convert to bf16 [N,K]
  transpose_w_kernel<<<dim3(kD / 32, 3 * kD / 32), dim3(256), 0, stream>>>(
      qkv_w, qkvWt, kD, 3 * kD);
  transpose_w_kernel<<<dim3(kD / 32, kD / 32), dim3(256), 0, stream>>>(
      proj_w, prjWt, kD, kD);
  transpose_w_kernel<<<dim3(kD / 32, kFF / 32), dim3(256), 0, stream>>>(
      ff1_w, ff1Wt, kD, kFF);
  transpose_w_kernel<<<dim3(kFF / 32, kD / 32), dim3(256), 0, stream>>>(
      ff2_w, ff2Wt, kFF, kD);

  // 1. LN1
  layernorm_kernel<<<dim3(M), dim3(256), 0, stream>>>(x, ln1_g, ln1_b, xn1);
  // 2. QKV projection: [M,D] x [D,3D]
  gemm_wmma<MODE_BF16><<<dim3(3 * kD / 128, M / 128), dim3(256), 0, stream>>>(
      xn1, qkvWt, qkv_b, nullptr, qkv, nullptr, M, 3 * kD, kD);
  // 3. RoPE + Q/K/V layout split
  rope_split_kernel<<<dim3((kB * kL * kH * 32) / 256), dim3(256), 0, stream>>>(
      qkv, Qb, Kb, Vt);
  // 4. Causal flash attention
  attn_kernel<<<dim3(kL / 128, kH, kB), dim3(256), 0, stream>>>(Qb, Kb, Vt, attn);
  // 5. Output projection + residual: x2 = x + attn @ proj_w + b
  gemm_wmma<MODE_RESID><<<dim3(kD / 128, M / 128), dim3(256), 0, stream>>>(
      attn, prjWt, proj_b, x, nullptr, x2, M, kD, kD);
  // 6. LN2
  layernorm_kernel<<<dim3(M), dim3(256), 0, stream>>>(x2, ln2_g, ln2_b, xn2);
  // 7. FF1 + exact GELU
  gemm_wmma<MODE_GELU><<<dim3(kFF / 128, M / 128), dim3(256), 0, stream>>>(
      xn2, ff1Wt, ff1_b, nullptr, hact, nullptr, M, kFF, kD);
  // 8. FF2 + residual -> final f32 output
  gemm_wmma<MODE_RESID><<<dim3(kD / 128, M / 128), dim3(256), 0, stream>>>(
      hact, ff2Wt, ff2_b, x2, nullptr, out, M, kD, kFF);
}
